// ComplexAttentionLayer_44873818309059
// MI455X (gfx1250) — matrix operand order repaired and checked by reading, not compiled, not run on hardware
//
#include <hip/hip_runtime.h>
#include <stdint.h>

// ComplexAttentionLayer for MI455X (gfx1250):
//  - all GEMMs on v_wmma_f32_16x16x32_bf16 (fp32 accumulate)
//  - causal block scheduling (score GEMM and PV GEMM only touch j <= i)
//  - K/V blocks staged into LDS by the Tensor Data Mover (TENSOR_LOAD_TO_LDS),
//    double-buffered on TENSORcnt, K fragments read from LDS as ds_load_b128, V as strided 16-bit reads.

namespace {
constexpr int B = 2, S = 4096, D = 128, H = 128;
constexpr int BS = B * S;
constexpr int NSH = B * S * H;          // elements per bf16 buffer
constexpr float EPS = 1e-6f;
constexpr float INV_SQRT_H = 0.08838834764831845f;  // 1/sqrt(128)
constexpr int BLK_ELEMS = 128 * H;      // one 128-row bf16 block = 16384 elems = 32KB
}

typedef __attribute__((ext_vector_type(16))) __bf16 v16bf;
typedef __attribute__((ext_vector_type(8)))  float  v8f;
typedef __attribute__((ext_vector_type(4)))  unsigned int tdm_v4u;
typedef __attribute__((ext_vector_type(8)))  int          tdm_v8i;
typedef __attribute__((ext_vector_type(4)))  int          tdm_v4i;

__device__ inline v8f wmma_bf16(v16bf a, v16bf b, v8f c) {
  return __builtin_amdgcn_wmma_f32_16x16x32_bf16(false, a, false, b, (short)0, c,
                                                 false, false);
}

__device__ inline v8f vzero8() { v8f z = {0.f,0.f,0.f,0.f,0.f,0.f,0.f,0.f}; return z; }

// ---------------------------------------------------------------------
// Tensor Data Mover: 1-D copy of `nelem` bf16 elements (contiguous) from
// global to LDS. D# per ISA 8.3-8.5: group0 {count=1, lds_addr, 57b global
// addr, type=2}; group1 {data_size=2B, tensor_dim0=nelem, dim1=1,
// tile_dim0=nelem, tile_dim1=1, dim0_stride=nelem}; groups 2/3 zero (<=2D).
// This toolchain exposes the 6-arg builtin (extra int32x8 group before cpol).
// ---------------------------------------------------------------------
__device__ inline void tdm_load_block(const void* lds_dst, const void* gsrc,
                                      unsigned nelem) {
  const unsigned long long ga = (unsigned long long)(uintptr_t)gsrc;
  tdm_v4u g0;
  g0[0] = 1u;                                              // count=1, user D#
  g0[1] = (unsigned)(uintptr_t)lds_dst;                    // lds_addr (bytes)
  g0[2] = (unsigned)ga;                                    // global_addr[31:0]
  g0[3] = (unsigned)((ga >> 32) & 0x01ffffffu) | 0x80000000u; // [56:32] | type=2
  tdm_v8i g1;
  g1[0] = (int)(1u << 16);                                 // data_size=1 (2B)
  g1[1] = (int)((nelem & 0xffffu) << 16);                  // tensor_dim0[15:0]
  g1[2] = (int)(((nelem >> 16) & 0xffffu) | (1u << 16));   // dim0[31:16], dim1=1
  g1[3] = (int)((nelem & 0xffffu) << 16);                  // tile_dim0 = nelem
  g1[4] = 1;                                               // tile_dim1 = 1
  g1[5] = (int)nelem;                                      // dim0_stride[31:0]
  g1[6] = 0;
  g1[7] = 0;
  const tdm_v4i z4 = {0, 0, 0, 0};
  const tdm_v8i z8 = {0, 0, 0, 0, 0, 0, 0, 0};
  __builtin_amdgcn_tensor_load_to_lds(g0, g1, z4, z4, z8, 0);
}

// ---- A fragment (16x32, 16-bit), contiguous k order used by this code: lane<16 row=lane K0..15,
// lane>=16 row=lane-16 K16..31. Two 16B chunks at +0 and +16 bytes.
__device__ inline v16bf load_a_f32(const float* __restrict__ src, int ld, int lane,
                                   float sgn) {
  const int half = (lane >> 4) & 1;
  const int r = lane & 15;
  const float* p = src + (size_t)r * ld + 16 * half;
  v16bf a;
#pragma unroll
  for (int v = 0; v < 8; ++v) {
    const int kb = 2 * v;
    a[2 * v]     = (__bf16)(sgn * p[kb]);
    a[2 * v + 1] = (__bf16)(sgn * p[kb + 1]);
  }
  return a;
}

__device__ inline v16bf load_a_bf16_v(const __bf16* __restrict__ src, int ld,
                                      int lane) {
  const int half = (lane >> 4) & 1;
  const int r = lane & 15;
  const __bf16* p = src + (size_t)r * ld + 16 * half;
  union { uint4 q[2]; v16bf v; } u;
  u.q[0] = *(const uint4*)(p);        // K chunk {0..7}+16*half
  u.q[1] = *(const uint4*)(p + 8);    // K chunk {8..15}+16*half
  return u.v;
}

__device__ inline v16bf load_a_bf16(const __bf16* __restrict__ src, int ld, int lane,
                                    float sgn) {
  const int half = (lane >> 4) & 1;
  const int r = lane & 15;
  const __bf16* p = src + (size_t)r * ld + 16 * half;
  v16bf a;
#pragma unroll
  for (int v = 0; v < 8; ++v) {
    const int kb = 2 * v;
    a[2 * v]     = (__bf16)(sgn * (float)p[kb]);
    a[2 * v + 1] = (__bf16)(sgn * (float)p[kb + 1]);
  }
  return a;
}

// ---- B fragment (32x16), same contiguous k order as A: lane<16 col=lane K0..15, lane>=16 K16..31.
// "Transposed source" B[k][n] = src[n*ld + k]: 16 consecutive elements/lane.
__device__ inline v16bf load_bT_f32(const float* __restrict__ src, int ld, int lane) {
  const int half = (lane >> 4) & 1;
  const int n = lane & 15;
  const float* p = src + (size_t)n * ld + 16 * half;
  v16bf b;
#pragma unroll
  for (int v = 0; v < 8; ++v) {
    b[2 * v]     = (__bf16)p[2 * v];
    b[2 * v + 1] = (__bf16)p[2 * v + 1];
  }
  return b;
}

// 16 contiguous bf16 at p (16B aligned) -> ds_load_b128 x2 when p is LDS.
__device__ inline v16bf load_b_contig(const __bf16* p) {
  union { uint4 q[2]; v16bf v; } u;
  u.q[0] = *(const uint4*)(p);
  u.q[1] = *(const uint4*)(p + 8);
  return u.v;
}

// Row-major source: B[k][n] = src[k*ld + n] (used for P @ V from LDS).
__device__ inline v16bf load_b_rm_bf16(const __bf16* __restrict__ src, int ld, int lane) {
  const int half = (lane >> 4) & 1;
  const int n = lane & 15;
  const __bf16* p = src + (size_t)(16 * half) * ld + n;
  v16bf b;
#pragma unroll
  for (int v = 0; v < 8; ++v) {
    b[2 * v]     = p[(size_t)(2 * v) * ld];
    b[2 * v + 1] = p[(size_t)(2 * v + 1) * ld];
  }
  return b;
}

// ---- C/D (16x16 f32): VGPR r -> row (r + 8*(lane>=16)), col = lane&15.
__device__ inline void store_c_bf16(__bf16* __restrict__ dst, int ld, int lane, v8f c) {
  const int half = (lane >> 4) & 1;
  const int n = lane & 15;
  __bf16* p = dst + (size_t)(8 * half) * ld + n;
#pragma unroll
  for (int r = 0; r < 8; ++r) p[(size_t)r * ld] = (__bf16)c[r];
}

// =====================================================================
// Kernel 1: complex linear projection (bf16 out).
// =====================================================================
__global__ void __launch_bounds__(256)
k_proj(const float* __restrict__ Z, const float* __restrict__ Wr,
       const float* __restrict__ Wi, const float* __restrict__ br,
       const float* __restrict__ bi, __bf16* __restrict__ Or,
       __bf16* __restrict__ Oi) {
  const int lane = threadIdx.x & 31, wv = threadIdx.x >> 5;
  const int rowslab = wv >> 1, colhalf = wv & 1;
  const int b = blockIdx.y;
  const int row0 = blockIdx.x * 64 + rowslab * 16;

  const float* xr = Z + ((size_t)(b * 2 + 0) * S + row0) * D;
  const float* xi = Z + ((size_t)(b * 2 + 1) * S + row0) * D;

  v8f accR[4], accI[4];
#pragma unroll
  for (int nt = 0; nt < 4; ++nt) { accR[nt] = vzero8(); accI[nt] = vzero8(); }

  for (int kc = 0; kc < D; kc += 32) {
    v16bf ar  = load_a_f32(xr + kc, D, lane,  1.f);
    v16bf ai  = load_a_f32(xi + kc, D, lane,  1.f);
    v16bf ain = load_a_f32(xi + kc, D, lane, -1.f);   // bf16 WMMA has no A-neg
#pragma unroll
    for (int nt = 0; nt < 4; ++nt) {
      const int n0 = colhalf * 64 + nt * 16;
      v16bf bwr = load_bT_f32(Wr + (size_t)n0 * D + kc, D, lane);
      v16bf bwi = load_bT_f32(Wi + (size_t)n0 * D + kc, D, lane);
      accR[nt] = wmma_bf16(ar,  bwr, accR[nt]);
      accR[nt] = wmma_bf16(ain, bwi, accR[nt]);
      accI[nt] = wmma_bf16(ai,  bwr, accI[nt]);
      accI[nt] = wmma_bf16(ar,  bwi, accI[nt]);
    }
  }

  const int n = lane & 15, half = (lane >> 4) & 1;
#pragma unroll
  for (int nt = 0; nt < 4; ++nt) {
    const int col = colhalf * 64 + nt * 16 + n;
    const float brv = br[col], biv = bi[col];
    const float bR = brv - biv, bI = brv + biv;
#pragma unroll
    for (int r = 0; r < 8; ++r) {
      const int row = row0 + half * 8 + r;
      Or[((size_t)b * S + row) * H + col] = (__bf16)(accR[nt][r] + bR);
      Oi[((size_t)b * S + row) * H + col] = (__bf16)(accI[nt][r] + bI);
    }
  }
}

// Row norms from the bf16-rounded Q/K (self-consistent with the WMMA QK^T).
__global__ void __launch_bounds__(256)
k_q2k2(const __bf16* __restrict__ Qr, const __bf16* __restrict__ Qi,
       const __bf16* __restrict__ Kr, const __bf16* __restrict__ Ki,
       float* __restrict__ q2, float* __restrict__ k2) {
  const int row = blockIdx.x * 256 + threadIdx.x;
  if (row >= BS) return;
  const __bf16* qr = Qr + (size_t)row * H;
  const __bf16* qi = Qi + (size_t)row * H;
  const __bf16* kr = Kr + (size_t)row * H;
  const __bf16* ki = Ki + (size_t)row * H;
  float aq = 0.f, ak = 0.f;
#pragma unroll 4
  for (int h = 0; h < H; ++h) {
    float a = (float)qr[h], c = (float)qi[h];
    float e = (float)kr[h], f = (float)ki[h];
    aq += a * a + c * c;
    ak += e * e + f * f;
  }
  q2[row] = aq;
  k2[row] = ak;
}

// =====================================================================
// Kernel 2a: causal score blocks. grid (S/128, B), block 256 (8 waves).
// K blocks (Kr,Ki: 32KB each, contiguous) are DMA'd to LDS by the TDM,
// double-buffered on TENSORcnt; only wave 0 issues/waits.
// Dynamic LDS: 2 bufs x (Kr+Ki) = 2*64KB = 128KB.
// =====================================================================
__global__ void __launch_bounds__(256)
k_scores(const __bf16* __restrict__ Qr, const __bf16* __restrict__ Qi,
         const __bf16* __restrict__ Kr, const __bf16* __restrict__ Ki,
         const float* __restrict__ q2, const float* __restrict__ k2,
         const float* __restrict__ nfp, const float* __restrict__ taup,
         float* __restrict__ attn) {
  extern __shared__ __align__(16) char smem[];
  __bf16* sbase = (__bf16*)smem;   // [buf][comp][BLK_ELEMS]

  const int lane = threadIdx.x & 31, wv = threadIdx.x >> 5;
  const int i = blockIdx.x, b = blockIdx.y;
  const int row0 = i * 128 + wv * 16;
  const int half = (lane >> 4) & 1, n = lane & 15;

  const float nf = nfp[0];
  const float nf2 = nf * nf;
  const float sp = log1pf(expf(taup[0]));

  float q2v[8];
#pragma unroll
  for (int r = 0; r < 8; ++r) q2v[r] = q2[(size_t)b * S + row0 + half * 8 + r];

  // Hoist Q fragments across the whole j loop (4 K-chunks, r & i).
  const __bf16* qrp = Qr + ((size_t)b * S + row0) * H;
  const __bf16* qip = Qi + ((size_t)b * S + row0) * H;
  v16bf aqr[4], aqi[4];
#pragma unroll
  for (int c4 = 0; c4 < 4; ++c4) {
    aqr[c4] = load_a_bf16_v(qrp + c4 * 32, H, lane);
    aqi[c4] = load_a_bf16_v(qip + c4 * 32, H, lane);
  }

  // Prologue: DMA block j=0 into buffer 0.
  if (wv == 0) {
    tdm_load_block(sbase,             Kr + ((size_t)b * S) * H, BLK_ELEMS);
    tdm_load_block(sbase + BLK_ELEMS, Ki + ((size_t)b * S) * H, BLK_ELEMS);
  }

  int cur = 0;
  for (int j = 0; j <= i; ++j) {
    if (wv == 0) {
      if (j < i) {  // kick off next block's DMA, then wait only for current
        const int nb = cur ^ 1;
        const size_t gofs = ((size_t)b * S + (size_t)(j + 1) * 128) * H;
        tdm_load_block(sbase + (size_t)nb * 2 * BLK_ELEMS,              Kr + gofs,
                       BLK_ELEMS);
        tdm_load_block(sbase + (size_t)nb * 2 * BLK_ELEMS + BLK_ELEMS, Ki + gofs,
                       BLK_ELEMS);
        __builtin_amdgcn_s_wait_tensorcnt(2);
      } else {
        __builtin_amdgcn_s_wait_tensorcnt(0);
      }
    }
    __syncthreads();   // LDS block `cur` is ready for all waves

    const __bf16* sKr = sbase + (size_t)cur * 2 * BLK_ELEMS;
    const __bf16* sKi = sKr + BLK_ELEMS;

    v8f acc[8];
#pragma unroll
    for (int nt = 0; nt < 8; ++nt) acc[nt] = vzero8();

#pragma unroll
    for (int c4 = 0; c4 < 4; ++c4) {
      const int kc = c4 * 32 + 16 * half;
#pragma unroll
      for (int nt = 0; nt < 8; ++nt) {
        const int krow = nt * 16 + n;               // key index within block
        acc[nt] = wmma_bf16(aqr[c4], load_b_contig(sKr + (size_t)krow * H + kc),
                            acc[nt]);
        acc[nt] = wmma_bf16(aqi[c4], load_b_contig(sKi + (size_t)krow * H + kc),
                            acc[nt]);
      }
    }

#pragma unroll
    for (int nt = 0; nt < 8; ++nt) {
      const int col = j * 128 + nt * 16 + n;
      const float k2v = k2[(size_t)b * S + col];
#pragma unroll
      for (int r = 0; r < 8; ++r) {
        const int row = row0 + half * 8 + r;
        float dd = nf2 + q2v[r] + k2v - 2.f * acc[nt][r] + EPS;
        float sc = -sp * logf(fmaxf(dd, 1e-12f)) * INV_SQRT_H;
        if (col > row) sc = -1e30f;
        attn[((size_t)b * S + row) * S + col] = sc;
      }
    }
    __syncthreads();   // all reads of buffer `cur` done before it is refilled
    cur ^= 1;
  }
}

// Kernel 2b: per-row online max / sum-of-exp over the causal extent.
__global__ void __launch_bounds__(256)
k_rowstats(const float* __restrict__ attn, float* __restrict__ mrow,
           float* __restrict__ lrow) {
  const int lane = threadIdx.x & 31, wv = threadIdx.x >> 5;
  const int s = blockIdx.x * 8 + wv;
  const int b = blockIdx.y;
  const float* rowp = attn + ((size_t)b * S + s) * S;

  float m = -INFINITY, l = 0.f;
  for (int t = lane; t <= s; t += 32) {
    float v = rowp[t];
    float nm = fmaxf(m, v);
    float e1 = (m == -INFINITY) ? 0.f : __expf(m - nm);
    l = l * e1 + __expf(v - nm);
    m = nm;
  }
#pragma unroll
  for (int off = 16; off > 0; off >>= 1) {
    float om = __shfl_xor(m, off, 32);
    float ol = __shfl_xor(l, off, 32);
    float nm = fmaxf(m, om);
    float e1 = (m  == -INFINITY) ? 0.f : __expf(m  - nm);
    float e2 = (om == -INFINITY) ? 0.f : __expf(om - nm);
    l = l * e1 + ol * e2;
    m = nm;
  }
  if (lane == 0) {
    mrow[(size_t)b * S + s] = m;
    lrow[(size_t)b * S + s] = l;
  }
}

// Kernel 2c: in-place normalize -> final attention weights (zeros above diag).
__global__ void __launch_bounds__(256)
k_weights(float* __restrict__ attn, const float* __restrict__ mrow,
          const float* __restrict__ lrow) {
  const int s = blockIdx.x, b = blockIdx.y;
  const float m = mrow[(size_t)b * S + s];
  const float inv = 1.f / lrow[(size_t)b * S + s];
  float* rowp = attn + ((size_t)b * S + s) * S;
  for (int t = threadIdx.x; t < S; t += 256) {
    float w = (t <= s) ? (__expf(rowp[t] - m) * inv) : 0.f;
    rowp[t] = w;
  }
}

// =====================================================================
// Kernel 2d: est = W @ V. grid (S/128, B), block 512 (16 waves).
// V blocks staged to LDS by TDM, double-buffered; causal K-extent.
// Dynamic LDS: 2 bufs x (Vr+Vi) = 128KB.
// =====================================================================
__global__ void __launch_bounds__(512)
k_est(const float* __restrict__ attn, const __bf16* __restrict__ Vr,
      const __bf16* __restrict__ Vi, __bf16* __restrict__ Er,
      __bf16* __restrict__ Ei) {
  extern __shared__ __align__(16) char smem[];
  __bf16* sbase = (__bf16*)smem;

  const int lane = threadIdx.x & 31, wv = threadIdx.x >> 5;
  const int rowslab = wv >> 1, colhalf = wv & 1;
  const int i = blockIdx.x, b = blockIdx.y;
  const int row0 = i * 128 + rowslab * 16;

  v8f accR[4], accI[4];
#pragma unroll
  for (int nt = 0; nt < 4; ++nt) { accR[nt] = vzero8(); accI[nt] = vzero8(); }

  const float* ap = attn + ((size_t)b * S + row0) * S;

  if (wv == 0) {
    tdm_load_block(sbase,             Vr + ((size_t)b * S) * H, BLK_ELEMS);
    tdm_load_block(sbase + BLK_ELEMS, Vi + ((size_t)b * S) * H, BLK_ELEMS);
  }

  int cur = 0;
  for (int tb = 0; tb <= i; ++tb) {
    if (wv == 0) {
      if (tb < i) {
        const int nb = cur ^ 1;
        const size_t gofs = ((size_t)b * S + (size_t)(tb + 1) * 128) * H;
        tdm_load_block(sbase + (size_t)nb * 2 * BLK_ELEMS,              Vr + gofs,
                       BLK_ELEMS);
        tdm_load_block(sbase + (size_t)nb * 2 * BLK_ELEMS + BLK_ELEMS, Vi + gofs,
                       BLK_ELEMS);
        __builtin_amdgcn_s_wait_tensorcnt(2);
      } else {
        __builtin_amdgcn_s_wait_tensorcnt(0);
      }
    }
    __syncthreads();

    const __bf16* sVr = sbase + (size_t)cur * 2 * BLK_ELEMS;
    const __bf16* sVi = sVr + BLK_ELEMS;

#pragma unroll
    for (int c4 = 0; c4 < 4; ++c4) {
      const int t = tb * 128 + c4 * 32;
      v16bf aw = load_a_f32(ap + t, S, lane, 1.f);  // fp32 weights -> bf16 A frag
#pragma unroll
      for (int nt = 0; nt < 4; ++nt) {
        const int n0 = colhalf * 64 + nt * 16;
        accR[nt] = wmma_bf16(aw, load_b_rm_bf16(sVr + (size_t)(c4 * 32) * H + n0,
                                                H, lane), accR[nt]);
        accI[nt] = wmma_bf16(aw, load_b_rm_bf16(sVi + (size_t)(c4 * 32) * H + n0,
                                                H, lane), accI[nt]);
      }
    }
    __syncthreads();
    cur ^= 1;
  }

#pragma unroll
  for (int nt = 0; nt < 4; ++nt) {
    const int n0 = colhalf * 64 + nt * 16;
    store_c_bf16(Er + ((size_t)b * S + row0) * H + n0, H, lane, accR[nt]);
    store_c_bf16(Ei + ((size_t)b * S + row0) * H + n0, H, lane, accI[nt]);
  }
}

// Kernel 3: output complex projection, fp32 into d_out (B,2,S,H,1).
__global__ void __launch_bounds__(256)
k_outproj(const __bf16* __restrict__ Er, const __bf16* __restrict__ Ei,
          const float* __restrict__ Wr, const float* __restrict__ Wi,
          const float* __restrict__ br, const float* __restrict__ bi,
          float* __restrict__ out) {
  const int lane = threadIdx.x & 31, wv = threadIdx.x >> 5;
  const int rowslab = wv >> 1, colhalf = wv & 1;
  const int b = blockIdx.y;
  const int row0 = blockIdx.x * 64 + rowslab * 16;

  const __bf16* er = Er + ((size_t)b * S + row0) * H;
  const __bf16* ei = Ei + ((size_t)b * S + row0) * H;

  v8f accR[4], accI[4];
#pragma unroll
  for (int nt = 0; nt < 4; ++nt) { accR[nt] = vzero8(); accI[nt] = vzero8(); }

  for (int kc = 0; kc < H; kc += 32) {
    v16bf ar  = load_a_bf16_v(er + kc, H, lane);
    v16bf ai  = load_a_bf16_v(ei + kc, H, lane);
    v16bf ain = load_a_bf16(ei + kc, H, lane, -1.f);  // compiler -> packed xor
#pragma unroll
    for (int nt = 0; nt < 4; ++nt) {
      const int n0 = colhalf * 64 + nt * 16;
      v16bf bwr = load_bT_f32(Wr + (size_t)n0 * D + kc, D, lane);
      v16bf bwi = load_bT_f32(Wi + (size_t)n0 * D + kc, D, lane);
      accR[nt] = wmma_bf16(ar,  bwr, accR[nt]);
      accR[nt] = wmma_bf16(ain, bwi, accR[nt]);
      accI[nt] = wmma_bf16(ai,  bwr, accI[nt]);
      accI[nt] = wmma_bf16(ar,  bwi, accI[nt]);
    }
  }

  const int n = lane & 15, half = (lane >> 4) & 1;
#pragma unroll
  for (int nt = 0; nt < 4; ++nt) {
    const int col = colhalf * 64 + nt * 16 + n;
    const float brv = br[col], biv = bi[col];
    const float bR = brv - biv, bI = brv + biv;
#pragma unroll
    for (int r = 0; r < 8; ++r) {
      const int row = row0 + half * 8 + r;
      out[(((size_t)b * 2 + 0) * S + row) * H + col] = accR[nt][r] + bR;
      out[(((size_t)b * 2 + 1) * S + row) * H + col] = accI[nt][r] + bI;
    }
  }
}

extern "C" void kernel_launch(void* const* d_in, const int* in_sizes, int n_in,
                              void* d_out, int out_size, void* d_ws, size_t ws_size,
                              hipStream_t stream) {
  (void)in_sizes; (void)n_in; (void)out_size; (void)ws_size;

  const float* Zq   = (const float*)d_in[0];
  const float* Zk   = (const float*)d_in[1];
  const float* Zv   = (const float*)d_in[2];
  const float* Wq_r = (const float*)d_in[3];
  const float* Wq_i = (const float*)d_in[4];
  const float* bq_r = (const float*)d_in[5];
  const float* bq_i = (const float*)d_in[6];
  const float* Wk_r = (const float*)d_in[7];
  const float* Wk_i = (const float*)d_in[8];
  const float* bk_r = (const float*)d_in[9];
  const float* bk_i = (const float*)d_in[10];
  const float* Wv_r = (const float*)d_in[11];
  const float* Wv_i = (const float*)d_in[12];
  const float* bv_r = (const float*)d_in[13];
  const float* bv_i = (const float*)d_in[14];
  const float* Wp_r = (const float*)d_in[15];
  const float* Wp_i = (const float*)d_in[16];
  const float* bp_r = (const float*)d_in[17];
  const float* bp_i = (const float*)d_in[18];
  const float* nf   = (const float*)d_in[19];
  const float* tau  = (const float*)d_in[20];

  float* out  = (float*)d_out;
  float* attn = out + (size_t)B * 2 * S * H;   // attn_weights region of d_out

  __bf16* Qr = (__bf16*)d_ws;
  __bf16* Qi = Qr + NSH;
  __bf16* Kr = Qi + NSH;
  __bf16* Ki = Kr + NSH;
  __bf16* Vr = Ki + NSH;
  __bf16* Vi = Vr + NSH;
  __bf16* Er = Vi + NSH;
  __bf16* Ei = Er + NSH;
  float* q2   = (float*)(Ei + NSH);
  float* k2   = q2 + BS;
  float* mrow = k2 + BS;
  float* lrow = mrow + BS;

  const size_t lds2buf = 2u * 2u * (size_t)BLK_ELEMS * sizeof(__bf16); // 128KB

  const dim3 gproj(S / 64, B);
  k_proj<<<gproj, 256, 0, stream>>>(Zq, Wq_r, Wq_i, bq_r, bq_i, Qr, Qi);
  k_proj<<<gproj, 256, 0, stream>>>(Zk, Wk_r, Wk_i, bk_r, bk_i, Kr, Ki);
  k_proj<<<gproj, 256, 0, stream>>>(Zv, Wv_r, Wv_i, bv_r, bv_i, Vr, Vi);

  k_q2k2<<<dim3(BS / 256), 256, 0, stream>>>(Qr, Qi, Kr, Ki, q2, k2);

  k_scores<<<dim3(S / 128, B), 256, lds2buf, stream>>>(Qr, Qi, Kr, Ki, q2, k2,
                                                       nf, tau, attn);
  k_rowstats<<<dim3(S / 8, B), 256, 0, stream>>>(attn, mrow, lrow);
  k_weights<<<dim3(S, B), 256, 0, stream>>>(attn, mrow, lrow);

  k_est<<<dim3(S / 128, B), 512, lds2buf, stream>>>(attn, Vr, Vi, Er, Ei);

  k_outproj<<<dim3(S / 64, B), 256, 0, stream>>>(Er, Ei, Wp_r, Wp_i, bp_r, bp_i, out);
}
